// RWA_18150531793695
// MI455X (gfx1250) — matrix operand order repaired and checked by reading, not compiled, not run on hardware
//
#include <hip/hip_runtime.h>

#define B_ 256
#define T_ 512
#define D_ 512
#define H_ 512

typedef __attribute__((ext_vector_type(4)))  float  v4f;
typedef __attribute__((ext_vector_type(4)))  __bf16 v4bf;
typedef __attribute__((ext_vector_type(8)))  __bf16 v8bf;
typedef __attribute__((ext_vector_type(16))) __bf16 v16bf;
typedef __attribute__((ext_vector_type(8)))  float  v8f;

// ---- helpers -------------------------------------------------------------

__device__ __forceinline__ __bf16 f2bf(float f) {
  union { float f; unsigned u; } v; v.f = f;
  unsigned r = v.u + 0x7FFFu + ((v.u >> 16) & 1u);   // round-to-nearest-even
  unsigned short s = (unsigned short)(r >> 16);
  return __builtin_bit_cast(__bf16, s);
}

// A-matrix 16x32 bf16 fragment (ISA 7.12.2): lane row = lane&15;
// half 0 (lanes 0-15): K = 0..7 and 16..23 ; half 1: K = 8..15 and 24..31.
__device__ __forceinline__ v16bf load_fragA(const __bf16* rowp, int halfSel) {
  const __bf16* p = rowp + halfSel * 8;
  v8bf lo = *(const v8bf*)(p);        // K = kb .. kb+7      (16B)
  v8bf hi = *(const v8bf*)(p + 16);   // K = kb+16 .. kb+23  (16B)
  return __builtin_shufflevector(lo, hi, 0,1,2,3,4,5,6,7,8,9,10,11,12,13,14,15);
}

// B-matrix 32x16 bf16 fragment: lane col = lane&15;
// half 0 holds K = 0..7 and 16..23, half 1 holds K = 8..15 and 24..31 (rowp is the
// n-major (transposed) weight row, so K is contiguous in memory).
__device__ __forceinline__ v16bf load_fragB(const __bf16* rowp, int halfSel) {
  const __bf16* p = rowp + halfSel * 8;
  v8bf lo = *(const v8bf*)(p);
  v8bf hi = *(const v8bf*)(p + 16);
  return __builtin_shufflevector(lo, hi, 0,1,2,3,4,5,6,7,8,9,10,11,12,13,14,15);
}

__device__ __forceinline__ v8f wmma_bf16(v16bf a, v16bf b, v8f c) {
  return __builtin_amdgcn_wmma_f32_16x16x32_bf16(
      /*neg_a=*/false, a, /*neg_b=*/false, b,
      /*c_mod=*/(short)0, c, /*reuse_a=*/false, /*reuse_b=*/false);
}

// Fragment array layout (C-fragment major, shared by producer & consumer):
//   F[t][wg (b/16)][tileN (n/16)][lane][slot]   slot r <-> row m = r + 8*(lane>=16)
__device__ __forceinline__ size_t fidx(int t, int wg, int tileN, int lane) {
  return ((((size_t)t * (B_ / 16) + wg) * (H_ / 16) + tileN) * 32 + lane) * 8;
}

// ---- kernel 1: x [B,T,D] f32 -> time-major bf16 xT [(t*B+b), D] ----------

__global__ void rwa_cvt_xT(const float* __restrict__ src,
                           __bf16* __restrict__ dst) {
  long i4 = ((long)blockIdx.x * blockDim.x + threadIdx.x) * 4;
  int g = (int)(i4 >> 9);          // source row b*T + t   (D_ == 512)
  int d = (int)(i4 & (D_ - 1));
  int b = g >> 9;                  // T_ == 512
  int t = g & (T_ - 1);
  v4f v = __builtin_nontemporal_load((const v4f*)(src + i4));   // single pass
  v4bf o; o.x = f2bf(v.x); o.y = f2bf(v.y); o.z = f2bf(v.z); o.w = f2bf(v.w);
  *(v4bf*)(dst + ((size_t)t * B_ + b) * D_ + d) = o;
}

// ---- kernel 2: transpose + convert weights: src[K,N] -> dst[N,K] bf16 ----

__global__ void rwa_transpose_cvt(const float* __restrict__ src,
                                  __bf16* __restrict__ dst, int K, int N) {
  long idx = (long)blockIdx.x * blockDim.x + threadIdx.x;
  if (idx >= (long)K * N) return;
  int k = (int)(idx / N);
  int nn = (int)(idx - (long)k * N);
  dst[(size_t)nn * K + k] = f2bf(__builtin_nontemporal_load(src + idx));
}

// ---- kernel 3: fused triple GEMM, 64x16 per wave -------------------------
// U  = x @ Wu + bu ; Gx = x @ Wg[:D] + bg ; Ax = x @ Wa[:D]
// A rows come from time-major xT, so each 16-row M-tile is (fixed t, 16
// consecutive b) == one scan fragment tile; accumulators are stored as
// contiguous per-lane v8f fragments with non-temporal hints (read-once
// stream, consumed much later by the scan).

__global__ __launch_bounds__(256)
void rwa_pre_gemm(const __bf16* __restrict__ xT,    // [T*B, D] time-major
                  const __bf16* __restrict__ WuT,   // [H, D] n-major
                  const __bf16* __restrict__ Wg1T,  // [H, D]
                  const __bf16* __restrict__ Wa1T,  // [H, D]
                  const float*  __restrict__ bu,
                  const float*  __restrict__ bg,
                  float* __restrict__ FU,           // fragment arrays
                  float* __restrict__ FG,
                  float* __restrict__ FA) {
  const int lane    = threadIdx.x & 31;
  const int wid     = blockIdx.x * (blockDim.x >> 5) + (threadIdx.x >> 5);
  const int tileN   = wid & 31;          // H/16 = 32 N-tiles
  const int mgroup  = wid >> 5;          // 2048 groups of 4 M-tiles
  const int halfSel = lane >> 4;
  const int lm      = lane & 15;
  const int n       = tileN * 16 + lm;

  const __bf16* aRow0 = xT + (size_t)(mgroup * 64 + lm) * D_;   // subtile s: +s*16 rows
  const __bf16* bU    = WuT  + (size_t)n * D_;
  const __bf16* bG    = Wg1T + (size_t)n * D_;
  const __bf16* bA    = Wa1T + (size_t)n * D_;

  const float bun = bu[n];
  const float bgn = bg[n];
  v8f accU[4], accG[4], accA[4];
#pragma unroll
  for (int s = 0; s < 4; ++s)
#pragma unroll
    for (int r = 0; r < 8; ++r) { accU[s][r] = bun; accG[s][r] = bgn; accA[s][r] = 0.f; }

  for (int k0 = 0; k0 < D_; k0 += 32) {
    v16bf wu = load_fragB(bU + k0, halfSel);
    v16bf wg = load_fragB(bG + k0, halfSel);
    v16bf wa = load_fragB(bA + k0, halfSel);
#pragma unroll
    for (int s = 0; s < 4; ++s) {
      v16bf a = load_fragA(aRow0 + (size_t)s * 16 * D_ + k0, halfSel);
      accU[s] = wmma_bf16(a, wu, accU[s]);
      accG[s] = wmma_bf16(a, wg, accG[s]);
      accA[s] = wmma_bf16(a, wa, accA[s]);
    }
  }

#pragma unroll
  for (int s = 0; s < 4; ++s) {
    int tileM = mgroup * 4 + s;
    int t  = tileM >> 4;                 // 16 M-tiles per time step
    int wg = tileM & 15;
    size_t fi = fidx(t, wg, tileN, lane);
    __builtin_nontemporal_store(accU[s], (v8f*)(FU + fi));
    __builtin_nontemporal_store(accG[s], (v8f*)(FG + fi));
    __builtin_nontemporal_store(accA[s], (v8f*)(FA + fi));
  }
}

// ---- kernel 4: persistent WMMA scan --------------------------------------
// One WG per 16-row batch slab (16 WGs). 32 waves; wave w owns N-tile w for
// all T steps. h lives in LDS (bf16); num/den/amax live in registers; the
// U/Gx/Ax fragment streams are software-pipelined one step ahead and loaded
// non-temporally so the cache hierarchy keeps the re-read weights instead.

__global__ __launch_bounds__(1024)
void rwa_scan(const float* __restrict__ FU,
              const float* __restrict__ FG,
              const float* __restrict__ FA,
              const __bf16* __restrict__ Wg2T,  // [H, H] n-major (Wg[D:])
              const __bf16* __restrict__ Wa2T,  // [H, H] n-major (Wa[D:])
              const float* __restrict__ s0,     // [H]
              float* __restrict__ out) {        // [B, T, H]
  __shared__ __bf16 hsh[16 * H_];               // 16 KB: h slab, bf16

  const int lane    = threadIdx.x & 31;
  const int wave    = threadIdx.x >> 5;         // == tileN, 0..31
  const int wg      = blockIdx.x;               // batch slab / 16
  const int rowBase = wg * 16;
  const int halfSel = lane >> 4;
  const int lm      = lane & 15;
  const int n       = wave * 16 + lm;           // this lane's fixed column

  const __bf16* bRowG = Wg2T + (size_t)n * H_;
  const __bf16* bRowA = Wa2T + (size_t)n * H_;

  // per-lane recurrent state: 8 elements at rows m = r + 8*halfSel, col n
  float num[8], den[8], amax[8];
#pragma unroll
  for (int r = 0; r < 8; ++r) { num[r] = 0.f; den[r] = 0.f; amax[r] = 1e-38f; }

  // h0 = tanh(s0) broadcast over batch
  const __bf16 h0b = f2bf(tanhf(s0[n]));
#pragma unroll
  for (int r = 0; r < 8; ++r) hsh[(size_t)(r + 8 * halfSel) * H_ + n] = h0b;
  __syncthreads();

  // prime the fragment pipeline (step 0)
  size_t fi0 = fidx(0, wg, wave, lane);
  v8f curG = __builtin_nontemporal_load((const v8f*)(FG + fi0));
  v8f curA = __builtin_nontemporal_load((const v8f*)(FA + fi0));
  v8f curU = __builtin_nontemporal_load((const v8f*)(FU + fi0));

  const __bf16* aRow = &hsh[(size_t)lm * H_];

  for (int t = 0; t < T_; ++t) {
    // issue next step's fragment loads now; consumed next iteration
    int tn = (t + 1 < T_) ? (t + 1) : t;
    size_t fin = fidx(tn, wg, wave, lane);
    v8f nxtG = __builtin_nontemporal_load((const v8f*)(FG + fin));
    v8f nxtA = __builtin_nontemporal_load((const v8f*)(FA + fin));
    v8f nxtU = __builtin_nontemporal_load((const v8f*)(FU + fin));

    v8f accG = curG;
    v8f accA = curA;

    // h @ Wg2 and h @ Wa2 : A from LDS, B streamed from cache-resident weights
#pragma unroll 4
    for (int k0 = 0; k0 < H_; k0 += 32) {
      v16bf a   = load_fragA(aRow + k0, halfSel);
      v16bf wgf = load_fragB(bRowG + k0, halfSel);
      v16bf waf = load_fragB(bRowA + k0, halfSel);
      accG = wmma_bf16(a, wgf, accG);
      accA = wmma_bf16(a, waf, accA);
    }

    // elementwise RWA update (state stays in registers)
    float hnew[8];
#pragma unroll
    for (int r = 0; r < 8; ++r) {
      int m = r + 8 * halfSel;
      float g  = tanhf(accG[r]);
      float z  = curU[r] * g;
      float a  = accA[r];
      float an = fmaxf(a, amax[r]);
      float e  = __expf(an - amax[r]);
      amax[r]  = an;
      num[r]  += z * e;
      den[r]  += e;
      float h  = tanhf(num[r] / den[r]);
      hnew[r]  = h;
      __builtin_nontemporal_store(
          h, out + ((size_t)(rowBase + m) * T_ + t) * H_ + n);
    }

    __syncthreads();                    // all hsh reads of step t done
#pragma unroll
    for (int r = 0; r < 8; ++r)
      hsh[(size_t)(r + 8 * halfSel) * H_ + n] = f2bf(hnew[r]);
    __syncthreads();                    // new h visible for step t+1

    curG = nxtG; curA = nxtA; curU = nxtU;
  }
}

// ---- host launch ---------------------------------------------------------

extern "C" void kernel_launch(void* const* d_in, const int* in_sizes, int n_in,
                              void* d_out, int out_size, void* d_ws, size_t ws_size,
                              hipStream_t stream) {
  (void)in_sizes; (void)n_in; (void)out_size; (void)ws_size;
  const float* x  = (const float*)d_in[0];
  const float* Wu = (const float*)d_in[1];
  const float* bu = (const float*)d_in[2];
  const float* Wg = (const float*)d_in[3];
  const float* bg = (const float*)d_in[4];
  const float* Wa = (const float*)d_in[5];
  const float* s0 = (const float*)d_in[6];
  float* out = (float*)d_out;

  const size_t BTH = (size_t)B_ * T_ * H_;     // 67,108,864
  const size_t BTD = (size_t)B_ * T_ * D_;

  // workspace carve (all chunks 16B+ aligned by construction)
  float*  FU   = (float*)d_ws;                  // fragment arrays, BTH floats each
  float*  FG   = FU + BTH;
  float*  FA   = FG + BTH;
  __bf16* xT   = (__bf16*)(FA + BTH);           // [T*B, D] time-major bf16
  __bf16* WuT  = xT   + BTD;                    // [H, D]
  __bf16* Wg1T = WuT  + (size_t)H_ * D_;        // [H, D]
  __bf16* Wg2T = Wg1T + (size_t)H_ * D_;        // [H, H]
  __bf16* Wa1T = Wg2T + (size_t)H_ * H_;        // [H, D]
  __bf16* Wa2T = Wa1T + (size_t)H_ * D_;        // [H, H]

  // 1) x -> time-major bf16 (exact multiple: BTD / (4*256) blocks)
  {
    int blocks = (int)(BTD / (4 * 256));
    rwa_cvt_xT<<<blocks, 256, 0, stream>>>(x, xT);
  }
  // 2) weight transpose+convert (Wg1 = first D rows of Wg, Wg2 = Wg + D*H)
  {
    int elems = D_ * H_;
    int blocks = (elems + 255) / 256;
    rwa_transpose_cvt<<<blocks, 256, 0, stream>>>(Wu, WuT, D_, H_);
    rwa_transpose_cvt<<<blocks, 256, 0, stream>>>(Wg, Wg1T, D_, H_);
    rwa_transpose_cvt<<<blocks, 256, 0, stream>>>(Wa, Wa1T, D_, H_);
    int elems2 = H_ * H_;
    int blocks2 = (elems2 + 255) / 256;
    rwa_transpose_cvt<<<blocks2, 256, 0, stream>>>(Wg + (size_t)D_ * H_, Wg2T, H_, H_);
    rwa_transpose_cvt<<<blocks2, 256, 0, stream>>>(Wa + (size_t)D_ * H_, Wa2T, H_, H_);
  }
  // 3) fused triple GEMM: 2048 M-groups (64 rows) x 32 N-tiles, 8 waves/block
  {
    int waves  = (B_ * T_ / 64) * (H_ / 16);    // 65536
    int blocks = waves / 8;                      // 8192
    rwa_pre_gemm<<<blocks, 256, 0, stream>>>(xT, WuT, Wg1T, Wa1T, bu, bg, FU, FG, FA);
  }
  // 4) persistent scan: one WG per 16-row batch slab
  {
    rwa_scan<<<B_ / 16, 1024, 0, stream>>>(FU, FG, FA, Wg2T, Wa2T, s0, out);
  }
}
